// HROMAttention_89824946029280
// MI455X (gfx1250) — matrix operand order repaired and checked by reading, not compiled, not run on hardware
//
#include <hip/hip_runtime.h>

// ---------------------------------------------------------------------------
// Types for WMMA
// ---------------------------------------------------------------------------
typedef __bf16 v16bf __attribute__((ext_vector_type(16)));
typedef float  v8f   __attribute__((ext_vector_type(8)));

#define WMMA_BF16(a, b, c) \
  __builtin_amdgcn_wmma_f32_16x16x32_bf16(false, (a), false, (b), (short)0, (c), false, false)

// Async copy 16 bytes global -> LDS (gfx1250 GLOBAL_LOAD_ASYNC_TO_LDS_B128).
// INST_OFFSET is applied to BOTH the LDS destination and the global source.
#define ASYNC_CP_B128(ldsoff, gaddr, imm)                                     \
  asm volatile("global_load_async_to_lds_b128 %0, %1, off offset:" #imm      \
               :: "v"(ldsoff), "v"(gaddr) : "memory")

#define WAIT_ASYNCCNT0() asm volatile("s_wait_asynccnt 0x0" ::: "memory")
#define WAIT_DSCNT0()    asm volatile("s_wait_dscnt 0x0" ::: "memory")

// ---------------------------------------------------------------------------
// Helpers
// ---------------------------------------------------------------------------
__device__ __forceinline__ unsigned short f2bf_u(float f) {
  unsigned u = __builtin_bit_cast(unsigned, f);
  unsigned r = u + 0x7FFFu + ((u >> 16) & 1u);   // round-to-nearest-even
  return (unsigned short)(r >> 16);
}
__device__ __forceinline__ __bf16 u2bf(unsigned short s) {
  return __builtin_bit_cast(__bf16, s);
}
__device__ __forceinline__ unsigned lds_off(const void* p) {
  // generic LDS pointer: low 32 bits are the LDS byte offset
  return (unsigned)(size_t)p;
}

// A-fragment (16x32 bf16), contiguous k layout:
//   lane l: row M = l&15, half = l>>4
//   VGPR pair p:  K = 2p + 16*half
__device__ __forceinline__ v16bf load_a_frag(const unsigned short* base, int stride, int lane) {
  int row = lane & 15, half = lane >> 4;
  const unsigned short* rp = base + (size_t)row * stride;
  v16bf a;
#pragma unroll
  for (int p = 0; p < 8; ++p) {
    int k = half * 16 + p * 2;
    unsigned u = *(const unsigned*)(rp + k);
    a[2 * p]     = u2bf((unsigned short)(u & 0xffffu));
    a[2 * p + 1] = u2bf((unsigned short)(u >> 16));
  }
  return a;
}

// B-fragment (32x16 bf16): lane l holds column N = l&15;
// element e -> K = e + 16*(l>>4). Memory layout "N-major": base[n*stride + k].
__device__ __forceinline__ v16bf load_b_frag(const unsigned short* base, int stride, int lane) {
  int n = lane & 15, half = lane >> 4;
  const unsigned short* np = base + (size_t)n * stride + half * 16;
  v16bf b;
#pragma unroll
  for (int q = 0; q < 8; ++q) {
    unsigned u = *(const unsigned*)(np + 2 * q);
    b[2 * q]     = u2bf((unsigned short)(u & 0xffffu));
    b[2 * q + 1] = u2bf((unsigned short)(u >> 16));
  }
  return b;
}

// B-fragment from K-major (row-major [k][n]) layout: base[k*stride + n].
__device__ __forceinline__ v16bf load_b_frag_kmajor(const unsigned short* base, int stride, int lane) {
  int n = lane & 15, half = lane >> 4;
  const unsigned short* p = base + (size_t)(half * 16) * stride + n;
  v16bf b;
#pragma unroll
  for (int e = 0; e < 16; ++e) b[e] = u2bf(p[(size_t)e * stride]);
  return b;
}

// ---------------------------------------------------------------------------
// fp32 -> bf16 conversion
// ---------------------------------------------------------------------------
__global__ void cvt_bf16_kernel(const float* __restrict__ in,
                                unsigned short* __restrict__ out, int n) {
  int i = blockIdx.x * blockDim.x + threadIdx.x;
  if (i < n) out[i] = f2bf_u(in[i]);
}

// ---------------------------------------------------------------------------
// Tiled bf16 GEMM with fp32 accumulate: C[M,N] = A[M,K] * B[K,N] + bias[N]
// block = 256 threads (8 waves). Block tile 128x64, wave tile 32x32, BK=32.
// A tile staged via async global->LDS copies (ASYNCcnt); B tile loaded
// coalesced (b128 per thread) and scattered transposed into LDS.
// ---------------------------------------------------------------------------
__global__ __launch_bounds__(256) void gemm_bf16_kernel(
    const unsigned short* __restrict__ A, const unsigned short* __restrict__ Bm,
    const float* __restrict__ bias, float* __restrict__ C, int M, int N, int K) {
  __shared__ unsigned short As[128 * 32];   // [row][k]
  __shared__ unsigned short Bts[64 * 32];   // transposed: [n][k]

  int tid  = threadIdx.x;
  int lane = tid & 31;
  int wave = tid >> 5;
  int m0 = blockIdx.y * 128;
  int n0 = blockIdx.x * 64;
  int wm = (wave & 3) * 32;
  int wn = (wave >> 2) * 32;

  v8f acc[2][2];
#pragma unroll
  for (int i = 0; i < 2; ++i)
#pragma unroll
    for (int j = 0; j < 2; ++j) acc[i][j] = (v8f){0.f, 0.f, 0.f, 0.f, 0.f, 0.f, 0.f, 0.f};

  // A staging: 2 threads per row, 16 elems (32B) each -> 2 async b128
  int arow = tid >> 1;
  int akk  = (tid & 1) * 16;
  const unsigned short* aptr = A + (size_t)(m0 + arow) * K + akk;

  // B staging: thread owns (k row, 8 consecutive n) -> one b128 load
  int bk = tid >> 3;            // 0..31
  int bn = (tid & 7) * 8;       // 0..56
  const unsigned short* bptr = Bm + (size_t)bk * N + n0 + bn;

  for (int k0 = 0; k0 < K; k0 += 32) {
    // ---- stage A tile (128x32) with async global->LDS b128 copies ----
    {
      unsigned long long ga = (unsigned long long)(size_t)(aptr + k0);
      unsigned la = lds_off(&As[arow * 32 + akk]);
      ASYNC_CP_B128(la, ga, 0);
      ASYNC_CP_B128(la, ga, 16);
    }
    // ---- stage B tile transposed: coalesced 16B load, scatter to LDS ----
    {
      uint4 u = *(const uint4*)(bptr + (size_t)k0 * N);
      unsigned short e[8];
      *(uint4*)e = u;
#pragma unroll
      for (int i = 0; i < 8; ++i) Bts[(bn + i) * 32 + bk] = e[i];
    }
    // prefetch next k-step tiles into cache
    if (k0 + 32 < K) {
      __builtin_prefetch(aptr + k0 + 32, 0, 1);
      __builtin_prefetch(bptr + (size_t)(k0 + 32) * N, 0, 1);
    }
    WAIT_ASYNCCNT0();       // A tile resident in LDS before barrier
    __syncthreads();

    v16bf a0 = load_a_frag(As + (wm + 0) * 32, 32, lane);
    v16bf a1 = load_a_frag(As + (wm + 16) * 32, 32, lane);
    v16bf b0 = load_b_frag(Bts + (wn + 0) * 32, 32, lane);
    v16bf b1 = load_b_frag(Bts + (wn + 16) * 32, 32, lane);

    acc[0][0] = WMMA_BF16(a0, b0, acc[0][0]);
    acc[0][1] = WMMA_BF16(a0, b1, acc[0][1]);
    acc[1][0] = WMMA_BF16(a1, b0, acc[1][0]);
    acc[1][1] = WMMA_BF16(a1, b1, acc[1][1]);

    __syncthreads();
  }

  // epilogue: C layout -> lane l: M = r + 8*(l>>4), N = l&15
  int half = lane >> 4, lc = lane & 15;
#pragma unroll
  for (int mi = 0; mi < 2; ++mi)
#pragma unroll
    for (int ni = 0; ni < 2; ++ni)
#pragma unroll
      for (int r = 0; r < 8; ++r) {
        int row = m0 + wm + mi * 16 + r + 8 * half;
        int col = n0 + wn + ni * 16 + lc;
        C[(size_t)row * N + col] = acc[mi][ni][r] + bias[col];
      }
}

// ---------------------------------------------------------------------------
// RoPE + QKV split: qkv fp32 [B,T,3,H,D] -> q,k (rotated), v bf16 [B,H,T,D]
// ---------------------------------------------------------------------------
__global__ void rope_split_kernel(const float* __restrict__ qkv,
                                  unsigned short* __restrict__ qb,
                                  unsigned short* __restrict__ kb,
                                  unsigned short* __restrict__ vb) {
  int idx = blockIdx.x * blockDim.x + threadIdx.x;     // b(1):h(4):t(11):d(6)
  int d = idx & 63;
  int t = (idx >> 6) & 2047;
  int h = (idx >> 17) & 15;
  int b = idx >> 21;

  size_t base = ((size_t)(b * 2048 + t)) * 3072 + h * 64 + d;
  float q = qkv[base];
  float k = qkv[base + 1024];
  float v = qkv[base + 2048];

  int i = d & 31;                                      // freq index
  float inv = __expf(-(float)i * 0.03125f * 9.210340371976184f);  // 10000^(-2i/64)
  float ang = (float)t * inv;
  float cs = __cosf(ang), sn = __sinf(ang);

  float qr = (d < 32) ? -qkv[base + 32]        : qkv[base - 32];
  float kr = (d < 32) ? -qkv[base + 1024 + 32] : qkv[base + 1024 - 32];

  qb[idx] = f2bf_u(q * cs + qr * sn);
  kb[idx] = f2bf_u(k * cs + kr * sn);
  vb[idx] = f2bf_u(v);
}

// ---------------------------------------------------------------------------
// Flash attention (causal): one wave per 16-row q tile, 32-key blocks.
// q,k,v: bf16 [B*H, T, 64]; out: fp32 [B, T, H*64]
// V tiles staged to LDS via async copies; P staged via ds stores.
// ---------------------------------------------------------------------------
__global__ __launch_bounds__(256) void attn_kernel(
    const unsigned short* __restrict__ qg, const unsigned short* __restrict__ kg,
    const unsigned short* __restrict__ vg, float* __restrict__ out) {
  constexpr int T = 2048, D = 64;
  __shared__ unsigned short Plds[8][16 * 32];  // per-wave probs, [row][key]
  __shared__ unsigned short Vr[8][32 * 64];    // per-wave V block, row-major [key][d]

  int tid  = threadIdx.x;
  int lane = tid & 31;
  int wave = tid >> 5;
  int half = lane >> 4, lc = lane & 15;
  int bh = blockIdx.y;
  int b = bh >> 4, h = bh & 15;
  int qt0 = blockIdx.x * 128 + wave * 16;

  const unsigned short* qbase = qg + (size_t)bh * T * D;
  const unsigned short* kbase = kg + (size_t)bh * T * D;
  const unsigned short* vbase = vg + (size_t)bh * T * D;

  // q A-fragments (row-major in global, stride D): d chunks [0,32) and [32,64)
  v16bf aq0 = load_a_frag(qbase + (size_t)qt0 * D, D, lane);
  v16bf aq1 = load_a_frag(qbase + (size_t)qt0 * D + 32, D, lane);

  float mrow[8], lrow[8], alpha[8];
  v8f o[4];
#pragma unroll
  for (int r = 0; r < 8; ++r) { mrow[r] = -3.0e38f; lrow[r] = 0.f; }
#pragma unroll
  for (int c = 0; c < 4; ++c) o[c] = (v8f){0.f, 0.f, 0.f, 0.f, 0.f, 0.f, 0.f, 0.f};

  for (int j0 = 0; j0 < qt0 + 16; j0 += 32) {
    // make sure prior ds_loads of Vr/Plds are done before async engine rewrites LDS
    WAIT_DSCNT0();
    // ---- stage V block (32 keys x 64 d, row-major) via async copies:
    //      lane owns one 128-byte V row -> 8 x b128
    {
      unsigned long long gv =
          (unsigned long long)(size_t)(vbase + (size_t)(j0 + lane) * D);
      unsigned lv = lds_off(&Vr[wave][lane * 64]);
      ASYNC_CP_B128(lv, gv, 0);
      ASYNC_CP_B128(lv, gv, 16);
      ASYNC_CP_B128(lv, gv, 32);
      ASYNC_CP_B128(lv, gv, 48);
      ASYNC_CP_B128(lv, gv, 64);
      ASYNC_CP_B128(lv, gv, 80);
      ASYNC_CP_B128(lv, gv, 96);
      ASYNC_CP_B128(lv, gv, 112);
    }

    // ---- scores S = Q K^T for 32 keys (two 16-key halves) ----
    v8f s0 = (v8f){0.f, 0.f, 0.f, 0.f, 0.f, 0.f, 0.f, 0.f};
    v8f s1 = s0;
    {
      v16bf bk;
      bk = load_b_frag(kbase + (size_t)j0 * D, D, lane);            s0 = WMMA_BF16(aq0, bk, s0);
      bk = load_b_frag(kbase + (size_t)j0 * D + 32, D, lane);       s0 = WMMA_BF16(aq1, bk, s0);
      bk = load_b_frag(kbase + (size_t)(j0 + 16) * D, D, lane);     s1 = WMMA_BF16(aq0, bk, s1);
      bk = load_b_frag(kbase + (size_t)(j0 + 16) * D + 32, D, lane);s1 = WMMA_BF16(aq1, bk, s1);
    }

    // ---- online softmax over the 32-key block ----
    int key0 = j0 + lc, key1 = key0 + 16;
#pragma unroll
    for (int r = 0; r < 8; ++r) {
      int qrow = qt0 + r + 8 * half;
      float v0 = s0[r] * 0.125f + ((key0 <= qrow) ? 0.f : -3.0e38f);
      float v1 = s1[r] * 0.125f + ((key1 <= qrow) ? 0.f : -3.0e38f);
      float mx = fmaxf(v0, v1);
      mx = fmaxf(mx, __shfl_xor(mx, 1, 32));
      mx = fmaxf(mx, __shfl_xor(mx, 2, 32));
      mx = fmaxf(mx, __shfl_xor(mx, 4, 32));
      mx = fmaxf(mx, __shfl_xor(mx, 8, 32));
      float mnew = fmaxf(mrow[r], mx);
      float a  = __expf(mrow[r] - mnew);
      float p0 = __expf(v0 - mnew);
      float p1 = __expf(v1 - mnew);
      float sum = p0 + p1;
      sum += __shfl_xor(sum, 1, 32);
      sum += __shfl_xor(sum, 2, 32);
      sum += __shfl_xor(sum, 4, 32);
      sum += __shfl_xor(sum, 8, 32);
      lrow[r] = lrow[r] * a + sum;
      mrow[r] = mnew;
      alpha[r] = a;
      Plds[wave][(r + 8 * half) * 32 + lc]      = f2bf_u(p0);
      Plds[wave][(r + 8 * half) * 32 + 16 + lc] = f2bf_u(p1);
    }
#pragma unroll
    for (int c = 0; c < 4; ++c)
#pragma unroll
      for (int r = 0; r < 8; ++r) o[c][r] *= alpha[r];

    // wave-local LDS handoff: P via ds pipe, V via async engine
    WAIT_DSCNT0();
    WAIT_ASYNCCNT0();

    // ---- O += P V ----
    v16bf ap = load_a_frag(&Plds[wave][0], 32, lane);
#pragma unroll
    for (int c = 0; c < 4; ++c) {
      v16bf bv = load_b_frag_kmajor(&Vr[wave][c * 16], 64, lane);
      o[c] = WMMA_BF16(ap, bv, o[c]);
    }
  }

  // ---- epilogue: normalize, write [B, T, H*64] ----
#pragma unroll
  for (int c = 0; c < 4; ++c)
#pragma unroll
    for (int r = 0; r < 8; ++r) {
      int t = qt0 + r + 8 * half;
      int col = h * 64 + c * 16 + lc;
      out[((size_t)b * T + t) * 1024 + col] = o[c][r] / lrow[r];
    }
}

// ---------------------------------------------------------------------------
// Host-side launch
// ---------------------------------------------------------------------------
extern "C" void kernel_launch(void* const* d_in, const int* in_sizes, int n_in,
                              void* d_out, int out_size, void* d_ws, size_t ws_size,
                              hipStream_t stream) {
  constexpr int B = 2, T = 2048, DIM = 1024, H = 16;
  constexpr int M = B * T;           // 4096
  constexpr int N3 = 3 * DIM;        // 3072
  constexpr size_t NE_X   = (size_t)M * DIM;        // 4,194,304
  constexpr size_t NE_WQ  = (size_t)DIM * N3;       // 3,145,728
  constexpr size_t NE_WP  = (size_t)DIM * DIM;      // 1,048,576
  constexpr size_t NE_QKV = (size_t)M * N3;
  constexpr size_t NE_HD  = (size_t)B * H * T * (DIM / H);  // == NE_X

  const float* x      = (const float*)d_in[0];
  // d_in[1] = additive causal mask (we apply causality analytically)
  const float* w_qkv  = (const float*)d_in[2];
  const float* b_qkv  = (const float*)d_in[3];
  const float* w_proj = (const float*)d_in[4];
  const float* b_proj = (const float*)d_in[5];
  float* outp = (float*)d_out;

  char* ws = (char*)d_ws;
  size_t off = 0;
  auto carve = [&](size_t bytes) -> void* {
    void* p = ws + off;
    off += (bytes + 255) & ~(size_t)255;
    return p;
  };
  unsigned short* xb     = (unsigned short*)carve(NE_X * 2);
  unsigned short* wqkvb  = (unsigned short*)carve(NE_WQ * 2);
  unsigned short* wprojb = (unsigned short*)carve(NE_WP * 2);
  float*          qkvf   = (float*)carve(NE_QKV * 4);
  unsigned short* qb     = (unsigned short*)carve(NE_HD * 2);
  unsigned short* kb     = (unsigned short*)carve(NE_HD * 2);
  unsigned short* vb     = (unsigned short*)carve(NE_HD * 2);
  float*          attnf  = (float*)carve(NE_X * 4);
  unsigned short* attnb  = (unsigned short*)carve(NE_X * 2);

  // 1) fp32 -> bf16 conversions
  cvt_bf16_kernel<<<(NE_X  + 255) / 256, 256, 0, stream>>>(x,      xb,     (int)NE_X);
  cvt_bf16_kernel<<<(NE_WQ + 255) / 256, 256, 0, stream>>>(w_qkv,  wqkvb,  (int)NE_WQ);
  cvt_bf16_kernel<<<(NE_WP + 255) / 256, 256, 0, stream>>>(w_proj, wprojb, (int)NE_WP);

  // 2) QKV projection: [4096,1024] x [1024,3072] + b_qkv
  gemm_bf16_kernel<<<dim3(N3 / 64, M / 128), 256, 0, stream>>>(
      xb, wqkvb, b_qkv, qkvf, M, N3, DIM);

  // 3) RoPE + split into q/k/v (bf16, [B,H,T,D])
  rope_split_kernel<<<(unsigned)(NE_HD / 256), 256, 0, stream>>>(qkvf, qb, kb, vb);

  // 4) causal flash attention -> attnf fp32 [B,T,DIM]
  attn_kernel<<<dim3(T / 128, B * H), 256, 0, stream>>>(qb, kb, vb, attnf);

  // 5) output projection
  cvt_bf16_kernel<<<(NE_X + 255) / 256, 256, 0, stream>>>(attnf, attnb, (int)NE_X);
  gemm_bf16_kernel<<<dim3(DIM / 64, M / 128), 256, 0, stream>>>(
      attnb, wprojb, b_proj, outp, M, DIM, DIM);
}
